// CrfLayer_35399120453848
// MI455X (gfx1250) — matrix operand order repaired and checked by reading, not compiled, not run on hardware
//
#include <hip/hip_runtime.h>

// CRF mean-field for H=W=128, C=2, 10 iterations on gfx1250 (MI455X).
//
// Bilateral path: dense 16384^2 Gaussian kernel K fused with the sm@K
// reduction via V_WMMA_F32_16X16X32_F16. A = [sm0; sm1; ones; 0...] (row 2
// yields bilateral_norm for free), B = freshly computed 32x16 f16 K-chunk.
// Each lane computes exactly the K values its B-slot needs (one column j,
// contiguous 16-run of i) -> no LDS transpose. K entries use the
// sq_i + sq_j - 2*dot form (7 VALU + 1 v_exp_f32 each, f32 so no
// cancellation issue); features pre-scaled by sqrt(0.5*log2 e) so the
// exponent is exp2-native. A-operand loads are branch-free via per-lane
// base pointers into {smh | ones | zeros}. Two j-tiles per wave amortize
// the 24 feature/sq loads and give two independent WMMA chains.

namespace {

constexpr int Hh = 128, Ww = 128, Cc = 2;
constexpr int Np = Hh * Ww;     // 16384
constexpr int NIT = 10;
constexpr float TA = 160.0f, TB = 3.0f, TG = 3.0f;

typedef __attribute__((ext_vector_type(16))) _Float16 v16h;
typedef __attribute__((ext_vector_type(8)))  _Float16 v8h;
typedef __attribute__((ext_vector_type(8)))  float    v8f;

// ---------- one-time setup ----------

// feats (SoA, pre-scaled by sqrt(0.5*log2 e)), sq = |f|^2, q = transpose(U)
__global__ __launch_bounds__(256)
void k_init(const float* __restrict__ un, const float* __restrict__ rgb,
            float* __restrict__ F, float* __restrict__ sq,
            float* __restrict__ q) {
  int n = blockIdx.x * blockDim.x + threadIdx.x;
  if (n >= Np) return;
  int y = n / Ww, x = n % Ww;
  const float s = 0.8493218002880191f;   // sqrt(0.5 * log2(e))
  float f0 = (float)x * (s / TA);
  float f1 = (float)y * (s / TA);
  float f2 = rgb[n * 3 + 0] * (s / TB);
  float f3 = rgb[n * 3 + 1] * (s / TB);
  float f4 = rgb[n * 3 + 2] * (s / TB);
  F[0 * Np + n] = f0; F[1 * Np + n] = f1; F[2 * Np + n] = f2;
  F[3 * Np + n] = f3; F[4 * Np + n] = f4;
  sq[n] = f0 * f0 + f1 * f1 + f2 * f2 + f3 * f3 + f4 * f4;
  q[0 * Np + n] = un[n * Cc + 0];
  q[1 * Np + n] = un[n * Cc + 1];
}

// 128x128 1-D Gaussian matrix, plus the A-operand constant rows buffer
__global__ __launch_bounds__(256)
void k_gauss(float* __restrict__ G, _Float16* __restrict__ cbuf) {
  int t = blockIdx.x * blockDim.x + threadIdx.x;
  if (t < 32) cbuf[t] = (t < 16) ? (_Float16)1.0f : (_Float16)0.0f;
  if (t >= Hh * Ww) return;
  int i = t >> 7, j = t & 127;
  float d = (float)(i - j) / TG;
  G[t] = __builtin_amdgcn_exp2f(-0.72134752044448170f * d * d); // exp(-d^2/2)
}

// row sums of G (spatial_norm[i][l] = rG[i]*rG[l])
__global__ __launch_bounds__(128)
void k_rowsum(const float* __restrict__ G, float* __restrict__ rG) {
  int i = threadIdx.x;
  float s = 0.0f;
  for (int j = 0; j < 128; ++j) s += G[i * 128 + j];
  rG[i] = s;
}

// ---------- per-iteration kernels ----------

__global__ __launch_bounds__(256)
void k_softmax(const float* __restrict__ q, float* __restrict__ sm,
               _Float16* __restrict__ smh) {
  int n = blockIdx.x * blockDim.x + threadIdx.x;
  if (n >= Np) return;
  float a = q[n], b = q[Np + n];
  float m = fmaxf(a, b);
  float ea = __builtin_amdgcn_exp2f((a - m) * 1.4426950408889634f);
  float eb = __builtin_amdgcn_exp2f((b - m) * 1.4426950408889634f);
  float inv = 1.0f / (ea + eb);
  float s0 = ea * inv, s1 = eb * inv;
  sm[n] = s0;        sm[Np + n] = s1;
  smh[n] = (_Float16)s0;  smh[Np + n] = (_Float16)s1;
}

// T[c,i,l] = sum_j G[i,j] * sm[c,j,l]
__global__ __launch_bounds__(256)
void k_matT(const float* __restrict__ G, const float* __restrict__ sm,
            float* __restrict__ T) {
  int t = blockIdx.x * blockDim.x + threadIdx.x;
  if (t >= Cc * Np) return;
  int c = t >> 14, i = (t >> 7) & 127, l = t & 127;
  const float* gr = G + i * 128;
  const float* sc = sm + c * Np + l;
  float acc = 0.0f;
  for (int j = 0; j < 128; ++j) acc = fmaf(gr[j], sc[j * 128], acc);
  T[t] = acc;
}

// sp[c,i,l] = (sum_k T[c,i,k] * G[k,l]) / (rG[i]*rG[l])
__global__ __launch_bounds__(256)
void k_matSp(const float* __restrict__ G, const float* __restrict__ T,
             const float* __restrict__ rG, float* __restrict__ sp) {
  int t = blockIdx.x * blockDim.x + threadIdx.x;
  if (t >= Cc * Np) return;
  int c = t >> 14, i = (t >> 7) & 127, l = t & 127;
  const float* tr = T + (c << 14) + i * 128;
  float acc = 0.0f;
  for (int k = 0; k < 128; ++k) acc = fmaf(tr[k], G[k * 128 + l], acc);
  sp[t] = acc / (rG[i] * rG[l]);
}

// bl[c,j] = (sum_i sm[c,i]*K[i,j]) / (sum_i K[i,j]); fused K generation.
// One wave per PAIR of adjacent 16-column j-tiles (32 columns / wave).
__global__ __launch_bounds__(256)
void k_bilateral(const float* __restrict__ F, const float* __restrict__ sq,
                 const _Float16* __restrict__ smh,
                 const _Float16* __restrict__ cbuf, float* __restrict__ bl) {
  int lane = threadIdx.x & 31;
  int wave = threadIdx.x >> 5;
  int pair = blockIdx.x * 8 + wave;   // 64 blocks * 8 waves = 512 pairs
  int j0   = pair * 32;
  int col  = lane & 15;               // B-operand column within tile
  int hi   = lane >> 4;               // half-wave selector
  int jA   = j0 + col;                // tile 0 column
  int jB   = j0 + 16 + col;           // tile 1 column

  // per-lane column features (pre-scaled) + negated squared norms
  float a0 = F[0 * Np + jA], a1 = F[1 * Np + jA], a2 = F[2 * Np + jA];
  float a3 = F[3 * Np + jA], a4 = F[4 * Np + jA];
  float b0 = F[0 * Np + jB], b1 = F[1 * Np + jB], b2 = F[2 * Np + jB];
  float b3 = F[3 * Np + jB], b4 = F[4 * Np + jB];
  float nsqA = -sq[jA], nsqB = -sq[jB];

  // Branch-free A-operand source: rows 0,1 -> smh; row 2 -> ones; else zeros.
  // A layout (matches B): lane L<16 covers K {0..15}; L>=16 covers {16..31}.
  const _Float16* abase;
  int astep, aoff2;
  if (col < 2)      { abase = smh + col * Np + hi * 16; astep = 32; aoff2 = 8; }
  else if (col == 2){ abase = cbuf;       astep = 0;  aoff2 = 8; }
  else              { abase = cbuf + 16;  astep = 0;  aoff2 = 8; }

  v8f accA = {}, accB = {};  // D rows: 0 -> bl[0], 1 -> bl[1], 2 -> norm

  for (int i0 = 0; i0 < Np; i0 += 32) {
    int ib = i0 + hi * 16;            // contiguous 16-run of i for this lane

    // 5x16 row features + 16 squared norms (16B-aligned float4 loads)
    float fi[5][16];
#pragma unroll
    for (int d = 0; d < 5; ++d) {
      const float4* p = (const float4*)(F + d * Np + ib);
#pragma unroll
      for (int t = 0; t < 4; ++t) {
        float4 v = p[t];
        fi[d][4 * t + 0] = v.x; fi[d][4 * t + 1] = v.y;
        fi[d][4 * t + 2] = v.z; fi[d][4 * t + 3] = v.w;
      }
    }
    float sqi[16];
#pragma unroll
    for (int t = 0; t < 4; ++t) {
      float4 v = ((const float4*)(sq + ib))[t];
      sqi[4 * t + 0] = v.x; sqi[4 * t + 1] = v.y;
      sqi[4 * t + 2] = v.z; sqi[4 * t + 3] = v.w;
    }

    // A operand: two unconditional 16B loads, no divergence
    v8h lo = *(const v8h*)abase;
    v8h h8 = *(const v8h*)(abase + aoff2);
    abase += astep;
    v16h A;
#pragma unroll
    for (int t = 0; t < 8; ++t) { A[t] = lo[t]; A[8 + t] = h8[t]; }

    // B operands: K = exp2(2*dot - sq_i - sq_j), 7 VALU + 1 exp per entry
    v16h Ba, Bb;
#pragma unroll
    for (int t = 0; t < 16; ++t) {
      float dA = fi[0][t] * a0;
      dA = fmaf(fi[1][t], a1, dA); dA = fmaf(fi[2][t], a2, dA);
      dA = fmaf(fi[3][t], a3, dA); dA = fmaf(fi[4][t], a4, dA);
      float eA = fmaf(2.0f, dA, nsqA - sqi[t]);
      Ba[t] = (_Float16)__builtin_amdgcn_exp2f(eA);

      float dB = fi[0][t] * b0;
      dB = fmaf(fi[1][t], b1, dB); dB = fmaf(fi[2][t], b2, dB);
      dB = fmaf(fi[3][t], b3, dB); dB = fmaf(fi[4][t], b4, dB);
      float eB = fmaf(2.0f, dB, nsqB - sqi[t]);
      Bb[t] = (_Float16)__builtin_amdgcn_exp2f(eB);
    }

    // two independent accumulator chains (fills WMMA->WMMA hazard window)
    accA = __builtin_amdgcn_wmma_f32_16x16x32_f16(
        false, A, false, Ba, (short)0, accA, false, false);
    accB = __builtin_amdgcn_wmma_f32_16x16x32_f16(
        false, A, false, Bb, (short)0, accB, false, false);
  }

  // D layout: lane n<16 holds (M=row, N=j+n) in v[row]
  if (lane < 16) {
    float invA = 1.0f / accA[2];
    bl[0 * Np + j0 + lane] = accA[0] * invA;
    bl[1 * Np + j0 + lane] = accA[1] * invA;
    float invB = 1.0f / accB[2];
    bl[0 * Np + j0 + 16 + lane] = accB[0] * invB;
    bl[1 * Np + j0 + 16 + lane] = accB[1] * invB;
  }
}

// q = U - comp @ (sw @ sp + bw @ bl)
__global__ __launch_bounds__(256)
void k_update(const float* __restrict__ un, const float* __restrict__ sp,
              const float* __restrict__ bl, const float* __restrict__ sw,
              const float* __restrict__ bw, const float* __restrict__ cm,
              float* __restrict__ q) {
  int n = blockIdx.x * blockDim.x + threadIdx.x;
  if (n >= Np) return;
  float s0 = sp[n], s1 = sp[Np + n], b0 = bl[n], b1 = bl[Np + n];
  float mp0 = sw[0] * s0 + sw[1] * s1 + bw[0] * b0 + bw[1] * b1;
  float mp1 = sw[2] * s0 + sw[3] * s1 + bw[2] * b0 + bw[3] * b1;
  float pw0 = cm[0] * mp0 + cm[1] * mp1;
  float pw1 = cm[2] * mp0 + cm[3] * mp1;
  q[n]      = un[n * Cc + 0] - pw0;
  q[Np + n] = un[n * Cc + 1] - pw1;
}

// output (1,H,W,C) from q (C,N)
__global__ __launch_bounds__(256)
void k_out(const float* __restrict__ q, float* __restrict__ out) {
  int n = blockIdx.x * blockDim.x + threadIdx.x;
  if (n >= Np) return;
  out[n * Cc + 0] = q[n];
  out[n * Cc + 1] = q[Np + n];
}

} // namespace

extern "C" void kernel_launch(void* const* d_in, const int* in_sizes, int n_in,
                              void* d_out, int out_size, void* d_ws, size_t ws_size,
                              hipStream_t stream) {
  const float* un  = (const float*)d_in[0];   // (1,H,W,C)
  const float* rgb = (const float*)d_in[1];   // (1,H,W,3)
  const float* sw  = (const float*)d_in[2];   // (C,C)
  const float* bw  = (const float*)d_in[3];   // (C,C)
  const float* cm  = (const float*)d_in[4];   // (C,C)
  float* out = (float*)d_out;

  // workspace layout (floats): ~1.2 MB total
  float* ws = (float*)d_ws;
  float* F    = ws;                    // 5*N feats (SoA, pre-scaled)
  float* sq   = F + 5 * Np;            // N squared norms
  float* G    = sq + Np;               // 128*128 gaussian
  float* rG   = G + Hh * Ww;           // 128 row sums
  _Float16* cbuf = (_Float16*)(rG + 128);     // 32 halves (ones | zeros)
  float* q    = rG + 128 + 16;         // 2*N
  float* sm   = q + 2 * Np;            // 2*N softmax f32
  _Float16* smh = (_Float16*)(sm + 2 * Np);   // 2*N halves (N floats)
  float* T    = sm + 2 * Np + Np;      // 2*N
  float* sp   = T + 2 * Np;            // 2*N
  float* bl   = sp + 2 * Np;           // 2*N

  const int nB  = Np / 256;            // 64
  const int nB2 = (Cc * Np) / 256;     // 128

  k_init<<<nB, 256, 0, stream>>>(un, rgb, F, sq, q);
  k_gauss<<<nB, 256, 0, stream>>>(G, cbuf);
  k_rowsum<<<1, 128, 0, stream>>>(G, rG);

  for (int it = 0; it < NIT; ++it) {
    k_softmax<<<nB, 256, 0, stream>>>(q, sm, smh);
    k_matT<<<nB2, 256, 0, stream>>>(G, sm, T);
    k_matSp<<<nB2, 256, 0, stream>>>(G, T, rG, sp);
    k_bilateral<<<64, 256, 0, stream>>>(F, sq, smh, cbuf, bl); // 512 tile-pairs
    k_update<<<nB, 256, 0, stream>>>(un, sp, bl, sw, bw, cm, q);
  }
  k_out<<<nB, 256, 0, stream>>>(q, out);
}
